// LegislativeGraphModel_34986803593677
// MI455X (gfx1250) — compile-verified, matrix operand order checked by reading, not run on hardware
//
#include <hip/hip_runtime.h>
#include <hip/hip_bf16.h>
#include <math.h>

// ---------------------------------------------------------------------------
// CDNA5 / gfx1250 types + WMMA helpers
// ---------------------------------------------------------------------------
typedef __bf16 bf16_t;
typedef __attribute__((ext_vector_type(16))) __bf16 v16bf;
typedef __attribute__((ext_vector_type(8)))  float  v8f;

#define DEV __device__ __forceinline__

DEV bf16_t f2bf(float f) {
  union { float f; unsigned u; } v; v.f = f;
  unsigned r = (v.u + 0x7FFFu + ((v.u >> 16) & 1u)) >> 16;  // round-nearest-even
  unsigned short s = (unsigned short)r;
  return __builtin_bit_cast(bf16_t, s);
}

// K index inside a 32-wide K-step for half-slot i of a lane (per ISA 16-bit A layout)
DEV int frag_k(int i, int hi) { return (i < 8 ? i : i + 8) + (hi ? 8 : 0); }

// inverse: (klocal, row m) -> (lane, half-slot i)
DEV void inv_frag(int klocal, int m, int& lane, int& i) {
  int hi, r8;
  if (klocal < 16) { hi = klocal >= 8; r8 = klocal - (hi ? 8 : 0); i = r8; }
  else { int kk = klocal - 16; hi = kk >= 8; r8 = kk - (hi ? 8 : 0); i = 8 + r8; }
  lane = m + (hi ? 16 : 0);
}

DEV v8f wmma_bf16(v16bf a, v16bf b, v8f c) {
  return __builtin_amdgcn_wmma_f32_16x16x32_bf16(false, a, false, b, (short)0, c,
                                                 false, false);
}

// ---------------------------------------------------------------------------
// Problem constants (from reference) and workspace layout (float offsets)
// ---------------------------------------------------------------------------
namespace cfg {
constexpr int H = 192, H4 = 48;
constexpr int ND = 100000, NLB = 30000, NLEG = 10000, NBV = 200000, NB = 80000;
constexpr int ED = 600000, EL = 400000, EVR = 200000, EVO = 400000;

constexpr size_t HACC_DON = 0;
constexpr size_t HACC_LOB = HACC_DON + (size_t)ND * H4;
constexpr size_t HACC_BV  = HACC_LOB + (size_t)NLB * H4;
constexpr size_t AGG_LD   = HACC_BV + (size_t)NBV * H4;
constexpr size_t AGG_LL   = AGG_LD + (size_t)NLEG * H;
constexpr size_t AGG_B    = AGG_LL + (size_t)NLEG * H;
constexpr size_t BVACC    = AGG_B + (size_t)NB * H;
constexpr size_t CNT_DON  = BVACC + (size_t)NBV * H;
constexpr size_t CNT_LOB  = CNT_DON + ND;
constexpr size_t CNT_BVT  = CNT_LOB + NLB;
constexpr size_t CNT_LD   = CNT_BVT + NBV;
constexpr size_t CNT_LL   = CNT_LD + NLEG;
constexpr size_t CNT_B    = CNT_LL + NLEG;
constexpr size_t CNT_BVV  = CNT_B + NB;
constexpr size_t ZERO_N   = CNT_BVV + NBV;
constexpr size_t WB_F     = (ZERO_N + 7) & ~(size_t)7;   // 32B aligned

// bf16 offsets inside WB region (fragment-major packed weights)
constexpr size_t WB_DON = 0;          // 384x192  -> 12kt*12tn*512
constexpr size_t WB_LOB = 73728;
constexpr size_t WB_VER = 147456;
constexpr size_t WB_V1  = 221184;     // vote layer1 (K=384)
constexpr size_t WB_V2  = 294912;     // vote layer2 (K=192) -> 6kt*12*512

// output tuple offsets (floats)
constexpr size_t O_DON  = 0;
constexpr size_t O_LOB  = (size_t)ND * H;
constexpr size_t O_LEG  = O_LOB + (size_t)NLB * H;
constexpr size_t O_BV   = O_LEG + (size_t)NLEG * H;
constexpr size_t O_BILL = O_BV + (size_t)NBV * H;
}  // namespace cfg

// ---------------------------------------------------------------------------
// Elementwise utility kernels
// ---------------------------------------------------------------------------
__global__ void zero_kernel(float* __restrict__ p, size_t n) {
  for (size_t i = (size_t)blockIdx.x * blockDim.x + threadIdx.x; i < n;
       i += (size_t)gridDim.x * blockDim.x)
    p[i] = 0.0f;
}

__global__ void copy_kernel(const float* __restrict__ s, float* __restrict__ d, size_t n) {
  for (size_t i = (size_t)blockIdx.x * blockDim.x + threadIdx.x; i < n;
       i += (size_t)gridDim.x * blockDim.x)
    d[i] = s[i];
}

// out = x + acc/max(cnt,1), row width 192
__global__ void residual_mean_kernel(const float* __restrict__ x, const float* __restrict__ acc,
                                     const float* __restrict__ cnt, float* __restrict__ out,
                                     size_t n) {
  for (size_t i = (size_t)blockIdx.x * blockDim.x + threadIdx.x; i < n;
       i += (size_t)gridDim.x * blockDim.x) {
    size_t node = i / 192;
    out[i] = x[i] + acc[i] / fmaxf(cnt[node], 1.0f);
  }
}

// ---------------------------------------------------------------------------
// Pack fp32 weight matrix W[192][Kw] (used transposed: B[k][n]=W[n][k]) into
// bf16 fragment-major layout: dst[(kt_base+kt)*12 + tn][lane][i]
// ---------------------------------------------------------------------------
__global__ void __launch_bounds__(256)
pack_frag_kernel(const float* __restrict__ W, int Kw, bf16_t* __restrict__ dst,
                 int kt_base, int num_kt) {
  int idx = blockIdx.x * 256 + threadIdx.x;
  int total = num_kt * 12 * 512;
  if (idx >= total) return;
  int kt  = idx / 6144;
  int rem = idx - kt * 6144;
  int tn  = rem / 512;
  int r2  = rem - tn * 512;
  int lane = r2 >> 4, i = r2 & 15;
  int k = kt * 32 + frag_k(i, lane >> 4);
  int n = tn * 16 + (lane & 15);
  dst[((size_t)(kt_base + kt) * 12 + tn) * 512 + r2] = f2bf(W[(size_t)n * Kw + k]);
}

// ---------------------------------------------------------------------------
// Temporal edge MLP, stage 1: per edge accumulate relu(W1*t+b1) (48-d hidden)
// onto its src node + edge count. (W2 is applied once per node later.)
// ---------------------------------------------------------------------------
__global__ void __launch_bounds__(256)
temporal_edge_kernel(const float* __restrict__ ts, const int* __restrict__ srci, int E,
                     const float* __restrict__ W1, const float* __restrict__ b1,
                     float* __restrict__ hacc, float* __restrict__ cnt) {
  __shared__ float w1s[48], b1s[48];
  if (threadIdx.x < 48) { w1s[threadIdx.x] = W1[threadIdx.x]; b1s[threadIdx.x] = b1[threadIdx.x]; }
  __syncthreads();
  size_t e = (size_t)blockIdx.x * 256 + threadIdx.x;
  if (e >= (size_t)E) return;
  float t = ts[e];
  int s = srci[e];
  float* hp = hacc + (size_t)s * 48;
#pragma unroll 8
  for (int j = 0; j < 48; ++j) {
    float h = fmaxf(fmaf(w1s[j], t, b1s[j]), 0.0f);
    atomicAdd(hp + j, h);
  }
  atomicAdd(cnt + s, 1.0f);
}

// ---------------------------------------------------------------------------
// Temporal stage 2: xs = x + (cnt>0 ? W2 @ mean_hidden + b2 : 0)
// 64 nodes per block; W2 (192x48) staged in LDS.
// ---------------------------------------------------------------------------
__global__ void __launch_bounds__(256)
inject_kernel(const float* __restrict__ x, const float* __restrict__ hacc,
              const float* __restrict__ cnt, const float* __restrict__ W2,
              const float* __restrict__ b2, float* __restrict__ xs, int N) {
  __shared__ float w2s[192 * 48];
  __shared__ float mhs[64 * 48];
  __shared__ float cc[64];
  const int tid = threadIdx.x;
  const size_t base = (size_t)blockIdx.x * 64;

  for (int idx = tid; idx < 192 * 48; idx += 256) w2s[idx] = W2[idx];
  for (int idx = tid; idx < 64 * 48; idx += 256) {
    size_t node = base + (size_t)(idx / 48);
    float v = 0.0f;
    if (node < (size_t)N) v = hacc[node * 48 + (idx % 48)] / fmaxf(cnt[node], 1.0f);
    mhs[idx] = v;
  }
  if (tid < 64) {
    size_t node = base + tid;
    cc[tid] = (node < (size_t)N) ? cnt[node] : 0.0f;
  }
  __syncthreads();

  for (int idx = tid; idx < 64 * 192; idx += 256) {
    int m = idx / 192, k = idx - m * 192;
    size_t node = base + m;
    if (node >= (size_t)N) continue;
    float o = 0.0f;
    if (cc[m] > 0.0f) {
      float acc = b2[k];
      const float* wr = w2s + k * 48;
      const float* mh = mhs + m * 48;
#pragma unroll 8
      for (int j = 0; j < 48; ++j) acc = fmaf(wr[j], mh[j], acc);
      o = acc;
    }
    xs[node * 192 + k] = x[node * 192 + k] + o;
  }
}

// ---------------------------------------------------------------------------
// SAGE aggregation: per-edge gather xs[src] -> atomic scatter-add onto agg[dst]
// One wave per edge (6 floats per lane).
// ---------------------------------------------------------------------------
__global__ void __launch_bounds__(256)
gather_kernel(const float* __restrict__ xs, const int* __restrict__ ei, int E,
              float* __restrict__ agg, float* __restrict__ cnt) {
  size_t e = (size_t)blockIdx.x * 8 + (threadIdx.x >> 5);
  int lane = threadIdx.x & 31;
  if (e >= (size_t)E) return;
  int s = ei[e], d = ei[(size_t)E + e];
  const float* sp = xs + (size_t)s * 192;
  float* dp = agg + (size_t)d * 192;
  for (int k = lane; k < 192; k += 32) atomicAdd(dp + k, sp[k]);
  if (lane == 0) atomicAdd(cnt + d, 1.0f);
}

// ---------------------------------------------------------------------------
// SAGE output: 16 dst nodes per block.
//   y = normalize( [agg/max(cnt,1) || x_dst] @ Wcat + (bl+br) )
//   mode 0: out = res + y     mode 1: out += y
// bf16 WMMA, K = 384 (12 k-steps), N = 192 (12 column tiles over 8 waves).
// ---------------------------------------------------------------------------
__global__ void __launch_bounds__(256)
sage_out_kernel(const float* __restrict__ agg, const float* __restrict__ cnt,
                const float* __restrict__ xd, const float* __restrict__ res,
                const bf16_t* __restrict__ WB, const float* __restrict__ bl,
                const float* __restrict__ br, float* __restrict__ out, int n_dst, int mode) {
  __shared__ v16bf AF[12 * 32];     // A fragments, fragment-major
  __shared__ float Ct[16 * 192];
  __shared__ float nrm[16];
  const int tid = threadIdx.x;
  const size_t base = (size_t)blockIdx.x * 16;

  bf16_t* AFb = (bf16_t*)AF;
  for (int idx = tid; idx < 12 * 512; idx += 256) {
    int kt = idx >> 9, rem = idx & 511;
    int lane = rem >> 4, i = rem & 15;
    int m = lane & 15, hi = lane >> 4;
    int k = kt * 32 + frag_k(i, hi);
    size_t node = base + m;
    float v = 0.0f;
    if (node < (size_t)n_dst) {
      if (k < 192) v = agg[node * 192 + k] / fmaxf(cnt[node], 1.0f);
      else         v = xd[node * 192 + (k - 192)];
    }
    AFb[idx] = f2bf(v);
  }
  __syncthreads();

  const int lane = tid & 31, wave = tid >> 5;
  for (int tn = wave; tn < 12; tn += 8) {
    v8f c = {0.f, 0.f, 0.f, 0.f, 0.f, 0.f, 0.f, 0.f};
#pragma unroll
    for (int kt = 0; kt < 12; ++kt) {
      v16bf a = AF[kt * 32 + lane];
      v16bf b = *(const v16bf*)(WB + ((size_t)(kt * 12 + tn) * 512 + lane * 16));
      c = wmma_bf16(a, b, c);
    }
    int col = tn * 16 + (lane & 15);
    int mb = (lane >> 4) ? 8 : 0;
#pragma unroll
    for (int r = 0; r < 8; ++r) Ct[(mb + r) * 192 + col] = c[r];
  }
  __syncthreads();

  if (tid < 16) {
    float ss = 0.0f;
    for (int n = 0; n < 192; ++n) {
      float v = Ct[tid * 192 + n] + bl[n] + br[n];
      Ct[tid * 192 + n] = v;
      ss = fmaf(v, v, ss);
    }
    nrm[tid] = fmaxf(sqrtf(ss), 1e-12f);
  }
  __syncthreads();

  for (int idx = tid; idx < 16 * 192; idx += 256) {
    int m = idx / 192, n = idx - m * 192;
    size_t node = base + m;
    if (node < (size_t)n_dst) {
      float y = Ct[idx] / nrm[m];
      size_t o = node * 192 + n;
      if (mode == 0) out[o] = res[o] + y;
      else           out[o] += y;
    }
  }
}

// ---------------------------------------------------------------------------
// Vote edge MLP: 16 edges/block. L1: 384->192 (relu), L2: 192->192, then
// scale by (clip(pol,0,1)+0.01) and atomic scatter-add onto bill-version dst.
// ---------------------------------------------------------------------------
__global__ void __launch_bounds__(256)
vote_kernel(const float* __restrict__ attr, const int* __restrict__ ei, int E,
            const bf16_t* __restrict__ W1f, const float* __restrict__ b1,
            const bf16_t* __restrict__ W2f, const float* __restrict__ b2,
            float* __restrict__ acc, float* __restrict__ cnt) {
  __shared__ v16bf AF1[12 * 32];
  __shared__ v16bf AF2[6 * 32];
  __shared__ float pols[16];
  __shared__ int   dstv[16];
  const int tid = threadIdx.x;
  const size_t base = (size_t)blockIdx.x * 16;

  if (tid < 16) {
    size_t e = base + tid;
    if (e < (size_t)E) {
      pols[tid] = fminf(fmaxf(attr[e * 385], 0.0f), 1.0f);
      int d = ei[(size_t)E + e];
      dstv[tid] = d;
      atomicAdd(cnt + d, 1.0f);
    } else { pols[tid] = 0.0f; dstv[tid] = 0; }
  }
  bf16_t* AF1b = (bf16_t*)AF1;
  for (int idx = tid; idx < 12 * 512; idx += 256) {
    int kt = idx >> 9, rem = idx & 511;
    int lane = rem >> 4, i = rem & 15;
    int m = lane & 15, hi = lane >> 4;
    int k = kt * 32 + frag_k(i, hi);
    size_t e = base + m;
    AF1b[idx] = f2bf((e < (size_t)E) ? attr[e * 385 + 1 + k] : 0.0f);
  }
  __syncthreads();

  const int lane = tid & 31, wave = tid >> 5;
  bf16_t* AF2b = (bf16_t*)AF2;

  // layer 1 + relu, write hidden straight into fragment-major LDS for layer 2
  for (int tn = wave; tn < 12; tn += 8) {
    v8f c = {0.f, 0.f, 0.f, 0.f, 0.f, 0.f, 0.f, 0.f};
#pragma unroll
    for (int kt = 0; kt < 12; ++kt) {
      v16bf a = AF1[kt * 32 + lane];
      v16bf b = *(const v16bf*)(W1f + ((size_t)(kt * 12 + tn) * 512 + lane * 16));
      c = wmma_bf16(a, b, c);
    }
    int col = tn * 16 + (lane & 15);
    int mb = (lane >> 4) ? 8 : 0;
#pragma unroll
    for (int r = 0; r < 8; ++r) {
      int m = mb + r;
      float h = fmaxf(c[r] + b1[col], 0.0f);
      int l2, i2;
      inv_frag(col & 31, m, l2, i2);
      AF2b[(size_t)(col >> 5) * 512 + l2 * 16 + i2] = f2bf(h);
    }
  }
  __syncthreads();

  // layer 2 + bias + polarity scale + atomic scatter
  for (int tn = wave; tn < 12; tn += 8) {
    v8f c = {0.f, 0.f, 0.f, 0.f, 0.f, 0.f, 0.f, 0.f};
#pragma unroll
    for (int kt = 0; kt < 6; ++kt) {
      v16bf a = AF2[kt * 32 + lane];
      v16bf b = *(const v16bf*)(W2f + ((size_t)(kt * 12 + tn) * 512 + lane * 16));
      c = wmma_bf16(a, b, c);
    }
    int col = tn * 16 + (lane & 15);
    int mb = (lane >> 4) ? 8 : 0;
#pragma unroll
    for (int r = 0; r < 8; ++r) {
      int m = mb + r;
      size_t e = base + m;
      if (e < (size_t)E) {
        float v = (c[r] + b2[col]) * (pols[m] + 0.01f);
        atomicAdd(acc + (size_t)dstv[m] * 192 + col, v);
      }
    }
  }
}

// ---------------------------------------------------------------------------
// Launch
// ---------------------------------------------------------------------------
extern "C" void kernel_launch(void* const* d_in, const int* in_sizes, int n_in,
                              void* d_out, int out_size, void* d_ws, size_t ws_size,
                              hipStream_t stream) {
  (void)in_sizes; (void)n_in; (void)out_size; (void)ws_size;
  using namespace cfg;

  const float* x_don = (const float*)d_in[0];
  const float* x_lob = (const float*)d_in[1];
  const float* x_leg = (const float*)d_in[2];
  const float* x_bv  = (const float*)d_in[3];
  const float* x_bil = (const float*)d_in[4];
  const int* ei_don = (const int*)d_in[5];
  const int* ei_lob = (const int*)d_in[6];
  const int* ei_ver = (const int*)d_in[7];
  const int* ei_vot = (const int*)d_in[8];
  const float* ts_don = (const float*)d_in[9];
  const float* ts_lob = (const float*)d_in[10];
  const float* ts_ver = (const float*)d_in[11];
  const float* vattr  = (const float*)d_in[12];
  const float* don_Wl = (const float*)d_in[13]; const float* don_bl = (const float*)d_in[14];
  const float* don_Wr = (const float*)d_in[15]; const float* don_br = (const float*)d_in[16];
  const float* lob_Wl = (const float*)d_in[17]; const float* lob_bl = (const float*)d_in[18];
  const float* lob_Wr = (const float*)d_in[19]; const float* lob_br = (const float*)d_in[20];
  const float* ver_Wl = (const float*)d_in[21]; const float* ver_bl = (const float*)d_in[22];
  const float* ver_Wr = (const float*)d_in[23]; const float* ver_br = (const float*)d_in[24];
  const float* tdon_W1 = (const float*)d_in[25]; const float* tdon_b1 = (const float*)d_in[26];
  const float* tdon_W2 = (const float*)d_in[27]; const float* tdon_b2 = (const float*)d_in[28];
  const float* tlob_W1 = (const float*)d_in[29]; const float* tlob_b1 = (const float*)d_in[30];
  const float* tlob_W2 = (const float*)d_in[31]; const float* tlob_b2 = (const float*)d_in[32];
  const float* tvot_W1 = (const float*)d_in[33]; const float* tvot_b1 = (const float*)d_in[34];
  const float* tvot_W2 = (const float*)d_in[35]; const float* tvot_b2 = (const float*)d_in[36];
  const float* vW1 = (const float*)d_in[37]; const float* vb1 = (const float*)d_in[38];
  const float* vW2 = (const float*)d_in[39]; const float* vb2 = (const float*)d_in[40];

  float* out = (float*)d_out;
  float* wsf = (float*)d_ws;
  bf16_t* wb = (bf16_t*)(wsf + WB_F);

  // xs scratch reuses output regions (overwritten with final values at the end)
  float* xs_don = out + O_DON;
  float* xs_lob = out + O_LOB;
  float* xs_bv  = out + O_BV;

  // 0) zero accumulators + counts
  zero_kernel<<<4096, 256, 0, stream>>>(wsf, ZERO_N);

  // 1) pack weights to bf16 fragment-major layout
  pack_frag_kernel<<<144, 256, 0, stream>>>(don_Wl, 192, wb + WB_DON, 0, 6);
  pack_frag_kernel<<<144, 256, 0, stream>>>(don_Wr, 192, wb + WB_DON, 6, 6);
  pack_frag_kernel<<<144, 256, 0, stream>>>(lob_Wl, 192, wb + WB_LOB, 0, 6);
  pack_frag_kernel<<<144, 256, 0, stream>>>(lob_Wr, 192, wb + WB_LOB, 6, 6);
  pack_frag_kernel<<<144, 256, 0, stream>>>(ver_Wl, 192, wb + WB_VER, 0, 6);
  pack_frag_kernel<<<144, 256, 0, stream>>>(ver_Wr, 192, wb + WB_VER, 6, 6);
  pack_frag_kernel<<<288, 256, 0, stream>>>(vW1, 384, wb + WB_V1, 0, 12);
  pack_frag_kernel<<<144, 256, 0, stream>>>(vW2, 192, wb + WB_V2, 0, 6);

  // 2) temporal hidden accumulation (48-d) per edge type
  temporal_edge_kernel<<<(ED + 255) / 256, 256, 0, stream>>>(
      ts_don, ei_don, ED, tdon_W1, tdon_b1, wsf + HACC_DON, wsf + CNT_DON);
  temporal_edge_kernel<<<(EL + 255) / 256, 256, 0, stream>>>(
      ts_lob, ei_lob, EL, tlob_W1, tlob_b1, wsf + HACC_LOB, wsf + CNT_LOB);
  temporal_edge_kernel<<<(EVR + 255) / 256, 256, 0, stream>>>(
      ts_ver, ei_ver, EVR, tvot_W1, tvot_b1, wsf + HACC_BV, wsf + CNT_BVT);

  // 3) xs = x + W2 @ mean_hidden + b2 (zero for nodes with no edges)
  inject_kernel<<<(ND + 63) / 64, 256, 0, stream>>>(
      x_don, wsf + HACC_DON, wsf + CNT_DON, tdon_W2, tdon_b2, xs_don, ND);
  inject_kernel<<<(NLB + 63) / 64, 256, 0, stream>>>(
      x_lob, wsf + HACC_LOB, wsf + CNT_LOB, tlob_W2, tlob_b2, xs_lob, NLB);
  inject_kernel<<<(NBV + 63) / 64, 256, 0, stream>>>(
      x_bv, wsf + HACC_BV, wsf + CNT_BVT, tvot_W2, tvot_b2, xs_bv, NBV);

  // 4) SAGE mean-aggregation (gather + atomic scatter)
  gather_kernel<<<(ED + 7) / 8, 256, 0, stream>>>(xs_don, ei_don, ED, wsf + AGG_LD, wsf + CNT_LD);
  gather_kernel<<<(EL + 7) / 8, 256, 0, stream>>>(xs_lob, ei_lob, EL, wsf + AGG_LL, wsf + CNT_LL);
  gather_kernel<<<(EVR + 7) / 8, 256, 0, stream>>>(xs_bv, ei_ver, EVR, wsf + AGG_B, wsf + CNT_B);

  // 5) SAGE linear + L2-normalize + residual (WMMA)
  sage_out_kernel<<<NLEG / 16, 256, 0, stream>>>(
      wsf + AGG_LD, wsf + CNT_LD, x_leg, x_leg, wb + WB_DON, don_bl, don_br,
      out + O_LEG, NLEG, 0);
  sage_out_kernel<<<NLEG / 16, 256, 0, stream>>>(
      wsf + AGG_LL, wsf + CNT_LL, x_leg, x_leg, wb + WB_LOB, lob_bl, lob_br,
      out + O_LEG, NLEG, 1);
  sage_out_kernel<<<NB / 16, 256, 0, stream>>>(
      wsf + AGG_B, wsf + CNT_B, x_bil, x_bil, wb + WB_VER, ver_bl, ver_br,
      out + O_BILL, NB, 0);

  // 6) vote-edge MLP (WMMA) -> scatter onto bill-version accumulator
  vote_kernel<<<(EVO + 15) / 16, 256, 0, stream>>>(
      vattr, ei_vot, EVO, wb + WB_V1, vb1, wb + WB_V2, vb2,
      wsf + BVACC, wsf + CNT_BVV);

  // 7) finalize outputs (overwrite xs scratch regions)
  residual_mean_kernel<<<4096, 256, 0, stream>>>(
      x_bv, wsf + BVACC, wsf + CNT_BVV, out + O_BV, (size_t)NBV * H);
  copy_kernel<<<4096, 256, 0, stream>>>(x_don, out + O_DON, (size_t)ND * H);
  copy_kernel<<<2048, 256, 0, stream>>>(x_lob, out + O_LOB, (size_t)NLB * H);
}
